// NoideAwareFeatureExtractor_46969762349488
// MI455X (gfx1250) — matrix operand order repaired and checked by reading, not compiled, not run on hardware
//
#include <hip/hip_runtime.h>
#include <hip/hip_bf16.h>
#include <math.h>

#define BATCH 256
#define NPTS  1000

typedef __attribute__((ext_vector_type(16))) _Float16 v16h;
typedef __attribute__((ext_vector_type(8)))  _Float16 v8h;
typedef __attribute__((ext_vector_type(8)))  float    v8f;

// ---------------------------------------------------------------------------
// Weight prep: W [Kin, Nout] f32  ->  WT [Nout, Kpad] f16 (zero-padded K rows)
// ---------------------------------------------------------------------------
__global__ void prep_weight_kernel(const float* __restrict__ W, _Float16* __restrict__ WT,
                                   int Kin, int Kpad, int Nout) {
    int gid = blockIdx.x * blockDim.x + threadIdx.x;
    int total = Kpad * Nout;
    if (gid >= total) return;
    int k = gid % Kpad;
    int n = gid / Kpad;
    WT[gid] = (k < Kin) ? (_Float16)W[(size_t)k * Nout + n] : (_Float16)0.0f;
}

// ---------------------------------------------------------------------------
// conf = sigmoid(x@Wc+bc); feat = [conf, x, 0-pad] as f16 [B,N,32]
// ---------------------------------------------------------------------------
__global__ void conf_feat_kernel(const float* __restrict__ x, const float* __restrict__ Wc,
                                 const float* __restrict__ bc, _Float16* __restrict__ feat) {
    size_t gid = (size_t)blockIdx.x * blockDim.x + threadIdx.x;
    if (gid >= (size_t)BATCH * NPTS) return;
    float x0 = x[gid * 3 + 0], x1 = x[gid * 3 + 1], x2 = x[gid * 3 + 2];
    float z = x0 * Wc[0] + x1 * Wc[1] + x2 * Wc[2] + bc[0];
    float conf = 1.0f / (1.0f + expf(-z));
    _Float16* o = feat + gid * 32;
    o[0] = (_Float16)conf; o[1] = (_Float16)x0; o[2] = (_Float16)x1; o[3] = (_Float16)x2;
#pragma unroll
    for (int c = 4; c < 32; ++c) o[c] = (_Float16)0.0f;
}

// ---------------------------------------------------------------------------
// WMMA GEMM: Y[M,Nc] = act(X[M,K] @ W[K,Nc] + bias), f16 in, f32 accum, f16 out
// X row-major [M,K] f16 (K % 32 == 0), WT row-major [Nc,K] f16 (transposed W).
// One wave computes a 16x32 output strip (two 16x16 N-tiles sharing the A
// fragment => 2 WMMAs per K-step, half the A traffic per WMMA). Nc % 32 == 0.
// ---------------------------------------------------------------------------
__global__ void __launch_bounds__(256) gemm_wmma_kernel(
    const _Float16* __restrict__ X, const _Float16* __restrict__ WT,
    const float* __restrict__ bias, _Float16* __restrict__ Y,
    int M, int K, int Nc, int relu) {
    const int lane  = threadIdx.x;            // 0..31
    const int wave  = threadIdx.y;            // 0..7
    const int tileM = blockIdx.x;             // M/16 tiles
    const int pair  = blockIdx.y * 8 + wave;  // 32-col strip index
    const int n0    = pair * 32;
    if (n0 >= Nc) return;                     // wave-uniform exit

    const int lmod = lane & 15;
    const int hsel = lane >> 4;               // 0: lanes 0-15, 1: lanes 16-31

    // A: row m = lmod; lane half selects K sub-chunks {0..7,16..23} / {8..15,24..31}
    const _Float16* arow  = X  + (size_t)(tileM * 16 + lmod) * K;
    // B: col n (row of WT); lane half selects K sub-chunks {0..7,16..23} / {8..15,24..31}
    const _Float16* brow0 = WT + (size_t)(n0 + lmod) * K;
    const _Float16* brow1 = WT + (size_t)(n0 + 16 + lmod) * K;

    v8f acc0 = {}, acc1 = {};
    for (int k0 = 0; k0 < K; k0 += 32) {
        const int ab = k0 + hsel * 8;
        v8h a_lo = *(const v8h*)(arow + ab);
        v8h a_hi = *(const v8h*)(arow + ab + 16);
        const int bb = k0 + hsel * 8;
        v8h b0_lo = *(const v8h*)(brow0 + bb);
        v8h b0_hi = *(const v8h*)(brow0 + bb + 16);
        v8h b1_lo = *(const v8h*)(brow1 + bb);
        v8h b1_hi = *(const v8h*)(brow1 + bb + 16);
        v16h a, b0, b1;
#pragma unroll
        for (int i = 0; i < 8; ++i) {
            a[i]  = a_lo[i];  a[i + 8]  = a_hi[i];
            b0[i] = b0_lo[i]; b0[i + 8] = b0_hi[i];
            b1[i] = b1_lo[i]; b1[i + 8] = b1_hi[i];
        }
        acc0 = __builtin_amdgcn_wmma_f32_16x16x32_f16(
            false, a, false, b0, (short)0, acc0, false, false);
        acc1 = __builtin_amdgcn_wmma_f32_16x16x32_f16(
            false, a, false, b1, (short)0, acc1, false, false);
    }

    // D layout: lanes 0-15 hold M=0..7 (VGPR r), lanes 16-31 hold M=8..15; N=lmod
    const float bn0 = bias[n0 + lmod];
    const float bn1 = bias[n0 + 16 + lmod];
#pragma unroll
    for (int r = 0; r < 8; ++r) {
        int m = hsel * 8 + r;
        float v0 = acc0[r] + bn0;
        float v1 = acc1[r] + bn1;
        if (relu) { v0 = fmaxf(v0, 0.0f); v1 = fmaxf(v1, 0.0f); }
        _Float16* yr = Y + (size_t)(tileM * 16 + m) * Nc + n0;
        yr[lmod]      = (_Float16)v0;
        yr[16 + lmod] = (_Float16)v1;
    }
}

// ---------------------------------------------------------------------------
// Linear interpolation along rows (align_corners=False), f16 channels
// ---------------------------------------------------------------------------
__global__ void interp_kernel(const _Float16* __restrict__ in, _Float16* __restrict__ out,
                              int Lin, int Lout, int C, size_t total) {
    size_t gid = (size_t)blockIdx.x * blockDim.x + threadIdx.x;
    if (gid >= total) return;
    int c = (int)(gid % C);
    size_t t = gid / C;
    int i = (int)(t % Lout);
    int b = (int)(t / Lout);
    float coord = ((float)i + 0.5f) * ((float)Lin / (float)Lout) - 0.5f;
    coord = fminf(fmaxf(coord, 0.0f), (float)(Lin - 1));
    int lo = (int)floorf(coord);
    int hi = (lo + 1 < Lin - 1) ? lo + 1 : Lin - 1;
    float w = coord - (float)lo;
    const _Float16* base = in + ((size_t)b * Lin) * C + c;
    float v = (float)base[(size_t)lo * C] * (1.0f - w) + (float)base[(size_t)hi * C] * w;
    out[gid] = (_Float16)v;
}

// ---------------------------------------------------------------------------
// Farthest point sampling, both problems fused in one launch:
//   blocks [0,256)  -> npoint=1000 with init1 -> idx1
//   blocks [256,512)-> npoint=500  with init2 -> idx2
// 128 threads (4 wave32), xyz + dist staged in LDS. Per-step argmax is a
// barrier-free wave32 __shfl_xor butterfly + one tiny 4-entry LDS combine
// (2 syncthreads/step instead of an 8-level LDS tree).
// ---------------------------------------------------------------------------
__global__ void __launch_bounds__(128) fps_kernel(const float* __restrict__ x,
                                                  const int* __restrict__ init1,
                                                  const int* __restrict__ init2,
                                                  int* __restrict__ idx1,
                                                  int* __restrict__ idx2) {
    const int job   = blockIdx.x;
    const int b     = job & (BATCH - 1);
    const int which = job >> 8;
    const int npoint   = which ? 500 : 1000;
    const int* init    = which ? init2 : init1;
    int*       idx_out = which ? idx2 : idx1;

    const int tid  = threadIdx.x;   // 0..127
    const int lane = tid & 31;
    const int wv   = tid >> 5;      // 0..3

    __shared__ float xs[NPTS], ys[NPTS], zs[NPTS], dist[NPTS];
    __shared__ float wbest[4];
    __shared__ int   wbesti[4];
    __shared__ int   s_far;

    const float* xb = x + (size_t)b * NPTS * 3;
    for (int i = tid; i < NPTS; i += 128) {
        xs[i] = xb[i * 3 + 0];
        ys[i] = xb[i * 3 + 1];
        zs[i] = xb[i * 3 + 2];
        dist[i] = 1e10f;
    }
    if (tid == 0) s_far = init[b];
    __syncthreads();

    for (int s = 0; s < npoint; ++s) {
        const int far = s_far;
        if (tid == 0) idx_out[(size_t)b * npoint + s] = far;
        const float cx = xs[far], cy = ys[far], cz = zs[far];
        float best = -1.0f; int besti = 0;
        for (int i = tid; i < NPTS; i += 128) {
            float dx = xs[i] - cx, dy = ys[i] - cy, dz = zs[i] - cz;
            float d = dx * dx + dy * dy + dz * dz;
            float dd = fminf(dist[i], d);
            dist[i] = dd;
            if (dd > best) { best = dd; besti = i; }
        }
        // wave32 butterfly argmax (smaller index wins ties)
#pragma unroll
        for (int off = 16; off > 0; off >>= 1) {
            float ob = __shfl_xor(best, off, 32);
            int   oi = __shfl_xor(besti, off, 32);
            if (ob > best || (ob == best && oi < besti)) { best = ob; besti = oi; }
        }
        if (lane == 0) { wbest[wv] = best; wbesti[wv] = besti; }
        __syncthreads();
        if (tid == 0) {
            float bb = wbest[0]; int bi = wbesti[0];
#pragma unroll
            for (int w = 1; w < 4; ++w) {
                if (wbest[w] > bb || (wbest[w] == bb && wbesti[w] < bi)) {
                    bb = wbest[w]; bi = wbesti[w];
                }
            }
            s_far = bi;
        }
        __syncthreads();
    }
}

// ---------------------------------------------------------------------------
// out[b,s,:] = [feat[b, idx[b,s], 0..Cin-1], x[b, idx[b,s], 0..2], zeros]
// ---------------------------------------------------------------------------
__global__ void gather_concat_kernel(const _Float16* __restrict__ feat,
                                     const float* __restrict__ x,
                                     const int* __restrict__ idx,
                                     _Float16* __restrict__ out,
                                     int S, int Cin, int Cout, size_t total) {
    size_t gid = (size_t)blockIdx.x * blockDim.x + threadIdx.x;
    if (gid >= total) return;
    int c = (int)(gid % Cout);
    size_t t = gid / Cout;
    int s = (int)(t % S);
    int b = (int)(t / S);
    int id = idx[(size_t)b * S + s];
    _Float16 v;
    if (c < Cin)          v = feat[((size_t)b * NPTS + id) * Cin + c];
    else if (c < Cin + 3) v = (_Float16)x[((size_t)b * NPTS + id) * 3 + (c - Cin)];
    else                  v = (_Float16)0.0f;
    out[gid] = v;
}

// ---------------------------------------------------------------------------
// Max over rows: L8 [B,250,512] f16 -> out [B,512] f32
// ---------------------------------------------------------------------------
__global__ void max_reduce_kernel(const _Float16* __restrict__ L8, float* __restrict__ out) {
    int gid = blockIdx.x * blockDim.x + threadIdx.x;
    if (gid >= BATCH * 512) return;
    int c = gid % 512;
    int b = gid / 512;
    const _Float16* p = L8 + ((size_t)b * 250) * 512 + c;
    float m = -1e30f;
    for (int r = 0; r < 250; ++r) m = fmaxf(m, (float)p[(size_t)r * 512]);
    out[gid] = m;
}

// ---------------------------------------------------------------------------
// Host orchestration
// ---------------------------------------------------------------------------
static inline size_t bump(size_t& off, size_t bytes) {
    size_t o = off;
    off = (off + bytes + 255) & ~(size_t)255;
    return o;
}

extern "C" void kernel_launch(void* const* d_in, const int* in_sizes, int n_in,
                              void* d_out, int out_size, void* d_ws, size_t ws_size,
                              hipStream_t stream) {
    (void)in_sizes; (void)n_in; (void)out_size; (void)ws_size;
    const float* x     = (const float*)d_in[0];
    const int*   init1 = (const int*)d_in[1];
    const int*   init2 = (const int*)d_in[2];
    const float* Wc  = (const float*)d_in[3];  const float* bc  = (const float*)d_in[4];
    const float* W1  = (const float*)d_in[5];  const float* b1  = (const float*)d_in[6];
    const float* W2  = (const float*)d_in[7];  const float* b2  = (const float*)d_in[8];
    const float* W3  = (const float*)d_in[9];  const float* b3  = (const float*)d_in[10];
    const float* W4  = (const float*)d_in[11]; const float* b4  = (const float*)d_in[12];
    const float* W5  = (const float*)d_in[13]; const float* b5  = (const float*)d_in[14];
    const float* W6  = (const float*)d_in[15]; const float* b6  = (const float*)d_in[16];
    const float* Wp1 = (const float*)d_in[17]; const float* bp1 = (const float*)d_in[18];
    const float* Wp2 = (const float*)d_in[19]; const float* bp2 = (const float*)d_in[20];
    float* out = (float*)d_out;
    char* ws = (char*)d_ws;

    // ---- workspace layout (f16 activations, buffer reuse) ----
    size_t off = 0;
    const size_t E = sizeof(_Float16);
    _Float16* feat = (_Float16*)(ws + bump(off, (size_t)BATCH * NPTS * 32 * E));
    _Float16* w1t  = (_Float16*)(ws + bump(off, (size_t)64  * 32  * E));
    _Float16* w2t  = (_Float16*)(ws + bump(off, (size_t)256 * 64  * E));
    _Float16* w3t  = (_Float16*)(ws + bump(off, (size_t)256 * 288 * E));
    _Float16* w4t  = (_Float16*)(ws + bump(off, (size_t)384 * 256 * E));
    _Float16* w5t  = (_Float16*)(ws + bump(off, (size_t)384 * 416 * E));
    _Float16* w6t  = (_Float16*)(ws + bump(off, (size_t)512 * 384 * E));
    _Float16* wp1t = (_Float16*)(ws + bump(off, (size_t)256 * 256 * E));
    _Float16* wp2t = (_Float16*)(ws + bump(off, (size_t)384 * 384 * E));
    _Float16* L1   = (_Float16*)(ws + bump(off, (size_t)BATCH * NPTS * 64 * E));
    _Float16* bufA = (_Float16*)(ws + bump(off, (size_t)BATCH * 1000 * 256 * E)); // L2 / U1 / L5 / NF2
    _Float16* bufB = (_Float16*)(ws + bump(off, (size_t)BATCH * 500  * 256 * E)); // D1 / L4 / DN2
    _Float16* bufC = (_Float16*)(ws + bump(off, (size_t)BATCH * 500  * 256 * E)); // L3 / D2 / L7
    _Float16* bufD = (_Float16*)(ws + bump(off, (size_t)BATCH * 500  * 288 * E)); // DN1 / L6 / L8
    _Float16* NF1  = (_Float16*)(ws + bump(off, (size_t)BATCH * 1000 * 288 * E));
    _Float16* U2   = (_Float16*)(ws + bump(off, (size_t)BATCH * 1000 * 384 * E));
    int* idx1 = (int*)(ws + bump(off, (size_t)BATCH * 1000 * sizeof(int)));
    int* idx2 = (int*)(ws + bump(off, (size_t)BATCH * 500  * sizeof(int)));

    auto eltGrid = [](size_t total) { return dim3((unsigned)((total + 255) / 256)); };
    const dim3 blk256(256);
    const dim3 gblk(32, 8);

    // ---- weight prep (f16, transposed, K-padded) ----
    prep_weight_kernel<<<eltGrid(64 * 32),   blk256, 0, stream>>>(W1,  w1t,  4,   32,  64);
    prep_weight_kernel<<<eltGrid(256 * 64),  blk256, 0, stream>>>(W2,  w2t,  64,  64,  256);
    prep_weight_kernel<<<eltGrid(256 * 288), blk256, 0, stream>>>(W3,  w3t,  259, 288, 256);
    prep_weight_kernel<<<eltGrid(384 * 256), blk256, 0, stream>>>(W4,  w4t,  256, 256, 384);
    prep_weight_kernel<<<eltGrid(384 * 416), blk256, 0, stream>>>(W5,  w5t,  387, 416, 384);
    prep_weight_kernel<<<eltGrid(512 * 384), blk256, 0, stream>>>(W6,  w6t,  384, 384, 512);
    prep_weight_kernel<<<eltGrid(256 * 256), blk256, 0, stream>>>(Wp1, wp1t, 256, 256, 256);
    prep_weight_kernel<<<eltGrid(384 * 384), blk256, 0, stream>>>(Wp2, wp2t, 384, 384, 384);

    // ---- both FPS problems in one launch (they are independent) ----
    fps_kernel<<<dim3(2 * BATCH), dim3(128), 0, stream>>>(x, init1, init2, idx1, idx2);

    // ---- conf + feat [B,1000,32] ----
    conf_feat_kernel<<<eltGrid((size_t)BATCH * NPTS), blk256, 0, stream>>>(x, Wc, bc, feat);

    // ---- layer1: [256000,32]x[32,64] relu ----
    gemm_wmma_kernel<<<dim3(256000 / 16, 1), gblk, 0, stream>>>(feat, w1t, b1, L1, 256000, 32, 64, 1);
    // ---- layer2: [256000,64]x[64,256] relu -> bufA ----
    gemm_wmma_kernel<<<dim3(256000 / 16, 1), gblk, 0, stream>>>(L1, w2t, b2, bufA, 256000, 64, 256, 1);
    // ---- interp 1000->500 -> bufB ----
    interp_kernel<<<eltGrid((size_t)BATCH * 500 * 256), blk256, 0, stream>>>(bufA, bufB, 1000, 500, 256, (size_t)BATCH * 500 * 256);
    // ---- layer3 (no relu): [128000,256]x[256,256] -> bufC ----
    gemm_wmma_kernel<<<dim3(128000 / 16, 1), gblk, 0, stream>>>(bufB, wp1t, bp1, bufC, 128000, 256, 256, 0);
    // ---- interp 500->1000 -> bufA (U1) ----
    interp_kernel<<<eltGrid((size_t)BATCH * 1000 * 256), blk256, 0, stream>>>(bufC, bufA, 500, 1000, 256, (size_t)BATCH * 1000 * 256);
    // ---- gather+concat via idx1 -> NF1 [B,1000,288] ----
    gather_concat_kernel<<<eltGrid((size_t)BATCH * 1000 * 288), blk256, 0, stream>>>(bufA, x, idx1, NF1, 1000, 256, 288, (size_t)BATCH * 1000 * 288);
    // ---- interp 1000->500 -> bufD (DN1 [B,500,288]) ----
    interp_kernel<<<eltGrid((size_t)BATCH * 500 * 288), blk256, 0, stream>>>(NF1, bufD, 1000, 500, 288, (size_t)BATCH * 500 * 288);
    // ---- layer4: [128000,288]x[288,256] relu -> bufB ----
    gemm_wmma_kernel<<<dim3(128000 / 16, 1), gblk, 0, stream>>>(bufD, w3t, b3, bufB, 128000, 288, 256, 1);
    // ---- layer5: [128000,256]x[256,384] relu -> bufA ----
    gemm_wmma_kernel<<<dim3(128000 / 16, 2), gblk, 0, stream>>>(bufB, w4t, b4, bufA, 128000, 256, 384, 1);
    // ---- interp 500->250 -> bufC (D2 [B,250,384]) ----
    interp_kernel<<<eltGrid((size_t)BATCH * 250 * 384), blk256, 0, stream>>>(bufA, bufC, 500, 250, 384, (size_t)BATCH * 250 * 384);
    // ---- layer6 (no relu): [64000,384]x[384,384] -> bufD ----
    gemm_wmma_kernel<<<dim3(64000 / 16, 2), gblk, 0, stream>>>(bufC, wp2t, bp2, bufD, 64000, 384, 384, 0);
    // ---- interp 250->1000 -> U2 [B,1000,384] ----
    interp_kernel<<<eltGrid((size_t)BATCH * 1000 * 384), blk256, 0, stream>>>(bufD, U2, 250, 1000, 384, (size_t)BATCH * 1000 * 384);
    // ---- gather+concat via idx2 -> bufA (NF2 [B,500,416]) ----
    gather_concat_kernel<<<eltGrid((size_t)BATCH * 500 * 416), blk256, 0, stream>>>(U2, x, idx2, bufA, 500, 384, 416, (size_t)BATCH * 500 * 416);
    // ---- interp 500->250 -> bufB (DN2 [B,250,416]) ----
    interp_kernel<<<eltGrid((size_t)BATCH * 250 * 416), blk256, 0, stream>>>(bufA, bufB, 500, 250, 416, (size_t)BATCH * 250 * 416);
    // ---- layer7: [64000,416]x[416,384] relu -> bufC ----
    gemm_wmma_kernel<<<dim3(64000 / 16, 2), gblk, 0, stream>>>(bufB, w5t, b5, bufC, 64000, 416, 384, 1);
    // (interp 250->250 is exact identity with align_corners=False: skipped)
    // ---- layer8: [64000,384]x[384,512] relu -> bufD ----
    gemm_wmma_kernel<<<dim3(64000 / 16, 2), gblk, 0, stream>>>(bufC, w6t, b6, bufD, 64000, 384, 512, 1);
    // ---- max over rows -> out [B,512] f32 ----
    max_reduce_kernel<<<eltGrid((size_t)BATCH * 512), blk256, 0, stream>>>(bufD, out);
}
